// ScaledDotProductAttention_40089224741525
// MI455X (gfx1250) — compile-verified, matrix operand order checked by reading, not run on hardware
//
#include <hip/hip_runtime.h>

#define B_   64
#define S_   1025
#define D_   768
#define H_   64
#define SP_  1056          // S padded to 33*32 key chunks / 66*16 tiles
#define SCALE_ 0.125f      // 1/sqrt(64)

typedef __attribute__((ext_vector_type(16))) __bf16 v16bf;
typedef __attribute__((ext_vector_type(8)))  __bf16 v8bf;
typedef __attribute__((ext_vector_type(8)))  float  v8f;

union FragBF { v16bf v; v8bf h[2]; __bf16 e[16]; };

// WMMA bf16 16-bit fragment K-index for element j in lane-half lh:
//   k = (j>>3)*16 + lh*8 + (j&7)   (ISA 7.12.2, 16-bit A 16x32 layout)

// ---------------------------------------------------------------------------
// Kernel 0: one-shot weight conversion f32 -> bf16 (wq pre-scaled by 1/8).
// ---------------------------------------------------------------------------
__global__ __launch_bounds__(256)
void sdpa_prep_weights(const float* __restrict__ wq,
                       const float* __restrict__ wk,
                       const float* __restrict__ wv,
                       __bf16* __restrict__ wqb,
                       __bf16* __restrict__ wkb,
                       __bf16* __restrict__ wvb) {
    const int i = blockIdx.x * 256 + threadIdx.x;
    if (i < H_ * D_) {
        wqb[i] = (__bf16)(wq[i] * SCALE_);
        wkb[i] = (__bf16)(wk[i]);
        wvb[i] = (__bf16)(wv[i]);
    }
}

// ---------------------------------------------------------------------------
// Kernel 1: Q/K/V projection.  One wave = 32 rows x 64 cols for q, k, v.
// Two A-row-tiles share every B (weight) fragment -> 24 WMMAs / K-chunk.
// v stored transposed (B,H,SP) so the flash kernel's P*V B-frags are linear.
// ---------------------------------------------------------------------------
__global__ __launch_bounds__(32)
void sdpa_qkv_proj(const float* __restrict__ x,
                   const __bf16* __restrict__ wqb,
                   const __bf16* __restrict__ wkb,
                   const __bf16* __restrict__ wvb,
                   __bf16* __restrict__ qb,
                   __bf16* __restrict__ kb,
                   __bf16* __restrict__ vt) {
    const int tile2 = blockIdx.x;         // 0..32  (32-row super-tiles)
    const int b     = blockIdx.y;         // 0..63
    const int lane  = threadIdx.x & 31;
    const int m     = lane & 15;          // A-row / B-col / D-col
    const int lh    = lane >> 4;          // lane half

    const int base = tile2 * 32;
    const int r0   = base + m;
    const int r1   = base + 16 + m;
    const float* xrow0 = x + ((size_t)b * S_ + (r0 < S_ ? r0 : S_ - 1)) * D_;
    const float* xrow1 = x + ((size_t)b * S_ + (r1 < S_ ? r1 : S_ - 1)) * D_;

    // acc[rowTile][matrix][nTile]
    v8f acc[2][3][4];
#pragma unroll
    for (int h = 0; h < 2; ++h)
#pragma unroll
        for (int w = 0; w < 3; ++w)
#pragma unroll
            for (int nt = 0; nt < 4; ++nt) {
                v8f z = {0.f,0.f,0.f,0.f,0.f,0.f,0.f,0.f};
                acc[h][w][nt] = z;
            }

    for (int kk = 0; kk < D_; kk += 32) {
        // ---- two A fragments: x rows, f32 -> bf16 ----
        FragBF a[2];
#pragma unroll
        for (int h = 0; h < 2; ++h) {
            const float* xr = h ? xrow1 : xrow0;
#pragma unroll
            for (int g = 0; g < 2; ++g) {
                const float4* p = reinterpret_cast<const float4*>(xr + kk + g*16 + lh*8);
                float4 f0 = p[0], f1 = p[1];
                a[h].e[g*8+0]=(__bf16)f0.x; a[h].e[g*8+1]=(__bf16)f0.y;
                a[h].e[g*8+2]=(__bf16)f0.z; a[h].e[g*8+3]=(__bf16)f0.w;
                a[h].e[g*8+4]=(__bf16)f1.x; a[h].e[g*8+5]=(__bf16)f1.y;
                a[h].e[g*8+6]=(__bf16)f1.z; a[h].e[g*8+7]=(__bf16)f1.w;
            }
        }
#pragma unroll
        for (int nt = 0; nt < 4; ++nt) {
            const int n = nt * 16 + m;     // output column (B-fragment lane col)
            const __bf16* wp0 = wqb + (size_t)n * D_ + kk;
            const __bf16* wp1 = wkb + (size_t)n * D_ + kk;
            const __bf16* wp2 = wvb + (size_t)n * D_ + kk;
#pragma unroll
            for (int w = 0; w < 3; ++w) {
                const __bf16* wp = (w == 0) ? wp0 : (w == 1) ? wp1 : wp2;
                FragBF bf;     // B fragment: direct 16-byte bf16 loads
                bf.h[0] = *(const v8bf*)(wp + lh*8);
                bf.h[1] = *(const v8bf*)(wp + 16 + lh*8);
                acc[0][w][nt] = __builtin_amdgcn_wmma_f32_16x16x32_bf16(
                    false, a[0].v, false, bf.v, (short)0, acc[0][w][nt], false, false);
                acc[1][w][nt] = __builtin_amdgcn_wmma_f32_16x16x32_bf16(
                    false, a[1].v, false, bf.v, (short)0, acc[1][w][nt], false, false);
            }
        }
    }

    // ---- store: D layout lane=(col m), VGPR r = row r + lh*8 ----
#pragma unroll
    for (int h = 0; h < 2; ++h)
#pragma unroll
        for (int nt = 0; nt < 4; ++nt)
#pragma unroll
            for (int r = 0; r < 8; ++r) {
                const int orow = base + h*16 + r + lh*8;
                const int n    = nt * 16 + m;
                const bool ok  = orow < S_;
                const size_t off = ((size_t)b * SP_ + orow) * H_ + n;
                qb[off] = ok ? (__bf16)(acc[h][0][nt][r]) : (__bf16)0.0f;
                kb[off] = ok ? (__bf16)(acc[h][1][nt][r]) : (__bf16)0.0f;
                vt[((size_t)b * H_ + n) * SP_ + orow] =
                          ok ? (__bf16)(acc[h][2][nt][r]) : (__bf16)0.0f;
            }
}

// ---------------------------------------------------------------------------
// Kernel 2: flash attention. One wave = 16 query rows, streams 32 keys/iter.
// ---------------------------------------------------------------------------
__global__ __launch_bounds__(32)
void sdpa_flash(const __bf16* __restrict__ qb,
                const __bf16* __restrict__ kb,
                const __bf16* __restrict__ vt,
                float* __restrict__ out) {
    __shared__ __bf16 pls[16 * 32] __attribute__((aligned(16)));

    const int qt   = blockIdx.x;          // 0..64 query tiles
    const int b    = blockIdx.y;
    const int lane = threadIdx.x & 31;
    const int m    = lane & 15;
    const int lh   = lane >> 4;

    // ---- preload Q A-fragments over d (two 32-wide chunks) ----
    const __bf16* qrow = qb + ((size_t)b * SP_ + qt * 16 + m) * H_;
    FragBF qf[2];
#pragma unroll
    for (int dc = 0; dc < 2; ++dc) {
        qf[dc].h[0] = *(const v8bf*)(qrow + dc*32 + lh*8);
        qf[dc].h[1] = *(const v8bf*)(qrow + dc*32 + 16 + lh*8);
    }

    float mr[8], lr[8];
    v8f ov[4];
#pragma unroll
    for (int r = 0; r < 8; ++r) { mr[r] = -1e30f; lr[r] = 0.f; }
#pragma unroll
    for (int dt = 0; dt < 4; ++dt) {
        v8f z = {0.f,0.f,0.f,0.f,0.f,0.f,0.f,0.f};
        ov[dt] = z;
    }

    for (int t0 = 0; t0 < SP_; t0 += 32) {
        if (t0 + 32 < SP_)
            __builtin_prefetch(kb + ((size_t)b * SP_ + t0 + 32 + m) * H_, 0, 1);

        // ---- scores: two 16-key tiles, accumulate over both d-chunks ----
        v8f sc[2];
#pragma unroll
        for (int nt = 0; nt < 2; ++nt) {
            v8f c = {0.f,0.f,0.f,0.f,0.f,0.f,0.f,0.f};
#pragma unroll
            for (int dc = 0; dc < 2; ++dc) {
                FragBF kf;  // B fragment of K^T: lane col = key, K-dim = d (contiguous)
                const __bf16* krow = kb + ((size_t)b * SP_ + t0 + nt*16 + m) * H_ + dc*32;
                kf.h[0] = *(const v8bf*)(krow + lh*8);
                kf.h[1] = *(const v8bf*)(krow + 16 + lh*8);
                c = __builtin_amdgcn_wmma_f32_16x16x32_bf16(false, qf[dc].v, false, kf.v, (short)0, c, false, false);
            }
            sc[nt] = c;
        }

        // ---- mask padded keys ----
#pragma unroll
        for (int nt = 0; nt < 2; ++nt) {
            if (t0 + nt*16 + m >= S_) {
#pragma unroll
                for (int r = 0; r < 8; ++r) sc[nt][r] = -1e30f;
            }
        }

        // ---- online softmax: row max / rescale / exp / row sum ----
        float alpha[8];
#pragma unroll
        for (int r = 0; r < 8; ++r) {
            float v = fmaxf(sc[0][r], sc[1][r]);
#pragma unroll
            for (int off = 1; off < 16; off <<= 1)
                v = fmaxf(v, __shfl_xor(v, off, 32));
            float nm = fmaxf(mr[r], v);
            alpha[r] = __expf(mr[r] - nm);
            mr[r]    = nm;
        }
#pragma unroll
        for (int nt = 0; nt < 2; ++nt)
#pragma unroll
            for (int r = 0; r < 8; ++r)
                sc[nt][r] = __expf(sc[nt][r] - mr[r]);
#pragma unroll
        for (int r = 0; r < 8; ++r) {
            float s = sc[0][r] + sc[1][r];
#pragma unroll
            for (int off = 1; off < 16; off <<= 1)
                s += __shfl_xor(s, off, 32);
            lr[r] = lr[r] * alpha[r] + s;
        }
#pragma unroll
        for (int dt = 0; dt < 4; ++dt)
#pragma unroll
            for (int r = 0; r < 8; ++r)
                ov[dt][r] *= alpha[r];

        // ---- P: C-layout -> A-fragment layout via LDS bounce ----
#pragma unroll
        for (int nt = 0; nt < 2; ++nt)
#pragma unroll
            for (int r = 0; r < 8; ++r)
                pls[(r + lh*8) * 32 + nt*16 + m] = (__bf16)sc[nt][r];
        __syncthreads();   // single-wave WG: S_NOP, ordering kept by DScnt
        FragBF pf;
        pf.h[0] = *(const v8bf*)(&pls[m*32 + lh*8]);
        pf.h[1] = *(const v8bf*)(&pls[m*32 + 16 + lh*8]);
        __syncthreads();

        // ---- P·V: V stored transposed so B fragments are contiguous ----
#pragma unroll
        for (int dt = 0; dt < 4; ++dt) {
            FragBF vf;
            const __bf16* vcol = vt + ((size_t)b * H_ + dt*16 + m) * SP_ + t0;
            vf.h[0] = *(const v8bf*)(vcol + lh*8);
            vf.h[1] = *(const v8bf*)(vcol + 16 + lh*8);
            ov[dt] = __builtin_amdgcn_wmma_f32_16x16x32_bf16(false, pf.v, false, vf.v, (short)0, ov[dt], false, false);
        }
    }

    // ---- normalize and store f32 output ----
#pragma unroll
    for (int dt = 0; dt < 4; ++dt)
#pragma unroll
        for (int r = 0; r < 8; ++r) {
            const int row = qt * 16 + r + lh * 8;
            if (row < S_)
                out[((size_t)b * S_ + row) * H_ + dt*16 + m] = ov[dt][r] / lr[r];
        }
}

extern "C" void kernel_launch(void* const* d_in, const int* in_sizes, int n_in,
                              void* d_out, int out_size, void* d_ws, size_t ws_size,
                              hipStream_t stream) {
    const float* x  = (const float*)d_in[0];
    const float* wq = (const float*)d_in[1];
    const float* wk = (const float*)d_in[2];
    const float* wv = (const float*)d_in[3];

    const size_t elems = (size_t)B_ * SP_ * H_;   // 4,325,376 bf16 each
    __bf16* qb  = (__bf16*)d_ws;
    __bf16* kb  = qb + elems;
    __bf16* vt  = kb + elems;                     // transposed (B,H,SP)
    __bf16* wqb = vt + elems;
    __bf16* wkb = wqb + (size_t)H_ * D_;
    __bf16* wvb = wkb + (size_t)H_ * D_;
    float*  out = (float*)d_out;

    sdpa_prep_weights<<<dim3((H_*D_ + 255)/256), 256, 0, stream>>>(wq, wk, wv, wqb, wkb, wvb);

    dim3 g1(SP_ / 32, B_);       // 33 x 64 super-tiles (32 rows each)
    sdpa_qkv_proj<<<g1, 32, 0, stream>>>(x, wqb, wkb, wvb, qb, kb, vt);

    dim3 g2((S_ + 15) / 16, B_); // 65 x 64 query tiles
    sdpa_flash<<<g2, 32, 0, stream>>>(qb, kb, vt, out);
}
